// AttentiveRNNLanguageModel_55783035240854
// MI455X (gfx1250) — matrix operand order repaired and checked by reading, not compiled, not run on hardware
//
#include <hip/hip_runtime.h>
#include <hip/hip_bf16.h>
#include <math.h>

typedef __attribute__((ext_vector_type(16))) _Float16 v16h;
typedef __attribute__((ext_vector_type(8)))  _Float16 v8h;
typedef __attribute__((ext_vector_type(8)))  float    v8f;
typedef __attribute__((ext_vector_type(4)))  unsigned int u32x4;
typedef __attribute__((ext_vector_type(8)))  int          i32x8;
typedef __attribute__((ext_vector_type(4)))  int          i32x4;
// GCC-style vector matching the async-builtin's pointee type exactly
typedef int b128_t __attribute__((vector_size(16)));

#define B_  4
#define T_  1024
#define H_  256
#define E_  256
#define G_  1024          // 4*H
#define V_  32000
#define K2_ 512           // 2*H

// ---- CDNA5 feature detection (device pass only) ---------------------------
#ifndef CDNA5_TRY_TDM
#define CDNA5_TRY_TDM 1
#endif
#ifndef CDNA5_TRY_ASYNC
#define CDNA5_TRY_ASYNC 1
#endif

#if defined(__AMDGCN__) && CDNA5_TRY_ASYNC && __has_builtin(__builtin_amdgcn_global_load_async_to_lds_b128)
#define HAVE_ASYNC 1
#else
#define HAVE_ASYNC 0
#endif
#if defined(__AMDGCN__) && CDNA5_TRY_TDM && __has_builtin(__builtin_amdgcn_tensor_load_to_lds)
#define HAVE_TDM 1
#else
#define HAVE_TDM 0
#endif

__device__ __forceinline__ void wait_async0() {
#if defined(__AMDGCN__)
#if __has_builtin(__builtin_amdgcn_s_wait_asynccnt)
    __builtin_amdgcn_s_wait_asynccnt(0);
#else
    asm volatile("s_wait_asynccnt 0x0" ::: "memory");
#endif
#endif
}
__device__ __forceinline__ void wait_tensor0() {
#if defined(__AMDGCN__)
#if __has_builtin(__builtin_amdgcn_s_wait_tensorcnt)
    __builtin_amdgcn_s_wait_tensorcnt(0);
#else
    asm volatile("s_wait_tensorcnt 0x0" ::: "memory");
#endif
#endif
}

#if HAVE_ASYNC
__device__ __forceinline__ void async_b128(const void* g, void* l) {
    __builtin_amdgcn_global_load_async_to_lds_b128(
        (__attribute__((address_space(1))) b128_t*)(uintptr_t)g,
        (__attribute__((address_space(3))) b128_t*)(uintptr_t)l,
        0, 0);
}
#endif

#if HAVE_TDM
// TDM 2D f16 tile load: tile1 rows x tile0 elems, row stride stride0 elems,
// LDS rows padded: 64B data + 16B pad (pad_interval=16 DW -> code 3, amount=4 DW -> code 3)
__device__ __forceinline__ void tdm_load_2d_f16(
    unsigned int lds_off, const _Float16* gptr,
    unsigned int dim0, unsigned int dim1,
    unsigned int tile0, unsigned int tile1, unsigned int stride0)
{
    unsigned long long ga = (unsigned long long)(uintptr_t)gptr;
    u32x4 g0;
    g0[0] = 1u;                                           // count=1, user descriptor
    g0[1] = lds_off;                                      // lds_addr (bytes)
    g0[2] = (unsigned int)ga;                             // global_addr[31:0]
    g0[3] = (unsigned int)((ga >> 32) & 0x1FFFFFFu)       // global_addr[56:32]
          | (2u << 30);                                   // type=2 (image)
    i32x8 g1;
    g1[0] = (int)((1u << 16)                              // data_size = 2B
                | (1u << 20)                              // pad_enable
                | (3u << 22)                              // pad_interval: 16 DWORDs
                | (3u << 25));                            // pad_amount: 4 DWORDs
    g1[1] = (int)((dim0 & 0xFFFFu) << 16);                // tensor_dim0[15:0]
    g1[2] = (int)(((dim0 >> 16) & 0xFFFFu) | ((dim1 & 0xFFFFu) << 16)); // dim0 hi | dim1 lo
    g1[3] = (int)(((dim1 >> 16) & 0xFFFFu) | (tile0 << 16));            // dim1 hi | tile_dim0
    g1[4] = (int)(tile1 & 0xFFFFu);                       // tile_dim1 (tile_dim2=0)
    g1[5] = (int)stride0;                                 // tensor_dim0_stride[31:0]
    g1[6] = 0;                                            // stride hi | dim1_stride lo
    g1[7] = 0;
    i32x4 z4 = {0, 0, 0, 0};
#if __clang_major__ >= 23
    i32x8 z8 = {0, 0, 0, 0, 0, 0, 0, 0};
    __builtin_amdgcn_tensor_load_to_lds(g0, g1, z4, z4, z8, 0);
#else
    __builtin_amdgcn_tensor_load_to_lds(g0, g1, z4, z4, 0);
#endif
}
#endif

// ---------------------------------------------------------------------------
// Generic f16 WMMA GEMM: C[M,N] = A[M,K] * Bw[N,K]^T + bias[N]  (act=1: tanh)
// Block tile 128x128, 8 waves, each wave 32x64 (2x4 tiles of 16x16).
// B tile staged via TDM (wave 0), A tile via per-lane async-to-LDS.
// ---------------------------------------------------------------------------
__global__ __launch_bounds__(256) void wmma_gemm_f16(
    const _Float16* __restrict__ A,    // [M,K] row-major, f16
    const _Float16* __restrict__ Bw,   // [N,K] row-major (i.e. B transposed)
    const float*    __restrict__ bias, // [N]
    float*          __restrict__ C,    // [M,N] row-major f32
    int M, int N, int K, int act)
{
    __shared__ _Float16 Asm[128][40];   // 32 K-halfs + pad 8 (80B row stride)
    __shared__ _Float16 Bsm[128][40];

    const int tid  = threadIdx.x;
    const int lane = tid & 31;
    const int wid  = tid >> 5;
    const int l16  = lane & 15;
    const int half = lane >> 4;

    const int mBlk = blockIdx.y * 128;
    const int nBlk = blockIdx.x * 128;
    const int wm   = (wid >> 1) * 32;   // 0,32,64,96
    const int wn   = (wid &  1) * 64;   // 0,64

    const v8f vzero = {0.f,0.f,0.f,0.f,0.f,0.f,0.f,0.f};
    v8f acc[2][4];
#pragma unroll
    for (int i = 0; i < 2; ++i)
#pragma unroll
        for (int j = 0; j < 4; ++j) acc[i][j] = vzero;

    // staging: 512 chunks of 8 halfs (16B); 2 chunks/thread
    const int c0  = tid, c1 = tid + 256;
    const int ar0 = c0 >> 2, ao0 = (c0 & 3) * 8;
    const int ar1 = c1 >> 2, ao1 = (c1 & 3) * 8;

    for (int k0 = 0; k0 < K; k0 += 32) {
        __syncthreads();   // previous compute done before LDS overwrite

#if HAVE_TDM
        if (wid == 0) {
            tdm_load_2d_f16((unsigned int)(uintptr_t)&Bsm[0][0],
                            &Bw[(size_t)nBlk * K + k0],
                            (unsigned int)(K - k0), (unsigned int)(N - nBlk),
                            32u, 128u, (unsigned int)K);
        }
#endif
#if HAVE_ASYNC
        async_b128(&A[(size_t)(mBlk + ar0) * K + k0 + ao0], &Asm[ar0][ao0]);
        async_b128(&A[(size_t)(mBlk + ar1) * K + k0 + ao1], &Asm[ar1][ao1]);
#if !HAVE_TDM
        async_b128(&Bw[(size_t)(nBlk + ar0) * K + k0 + ao0], &Bsm[ar0][ao0]);
        async_b128(&Bw[(size_t)(nBlk + ar1) * K + k0 + ao1], &Bsm[ar1][ao1]);
#endif
        wait_async0();
#else
        {   // de-serialized fallback: all loads first, then all stores
            v8h ta0 = *(const v8h*)&A[(size_t)(mBlk + ar0) * K + k0 + ao0];
            v8h ta1 = *(const v8h*)&A[(size_t)(mBlk + ar1) * K + k0 + ao1];
#if !HAVE_TDM
            v8h tb0 = *(const v8h*)&Bw[(size_t)(nBlk + ar0) * K + k0 + ao0];
            v8h tb1 = *(const v8h*)&Bw[(size_t)(nBlk + ar1) * K + k0 + ao1];
#endif
            if (k0 + 32 < K) {
                __builtin_prefetch(&A [(size_t)(mBlk + ar0) * K + k0 + 32 + ao0], 0, 1);
                __builtin_prefetch(&Bw[(size_t)(nBlk + ar0) * K + k0 + 32 + ao0], 0, 1);
            }
            *(v8h*)&Asm[ar0][ao0] = ta0;
            *(v8h*)&Asm[ar1][ao1] = ta1;
#if !HAVE_TDM
            *(v8h*)&Bsm[ar0][ao0] = tb0;
            *(v8h*)&Bsm[ar1][ao1] = tb1;
#endif
        }
#endif
#if HAVE_TDM
        if (wid == 0) wait_tensor0();
#endif
        __syncthreads();

        // A frag (ISA 16x32 f16 layout): lanes0-15 K{0-7,16-23}, lanes16-31 K{8-15,24-31}
        v16h af[2];
#pragma unroll
        for (int mt = 0; mt < 2; ++mt) {
            v8h lo = *(const v8h*)&Asm[wm + mt*16 + l16][half * 8];
            v8h hi = *(const v8h*)&Asm[wm + mt*16 + l16][16 + half * 8];
            af[mt] = __builtin_shufflevector(lo, hi, 0,1,2,3,4,5,6,7,8,9,10,11,12,13,14,15);
        }
        // B frag (32x16): col = lane%16, same K layout as A: K{0-7,16-23} / K{8-15,24-31}
        v16h bf[4];
#pragma unroll
        for (int nt = 0; nt < 4; ++nt) {
            v8h lo = *(const v8h*)&Bsm[wn + nt*16 + l16][half * 8];
            v8h hi = *(const v8h*)&Bsm[wn + nt*16 + l16][16 + half * 8];
            bf[nt] = __builtin_shufflevector(lo, hi, 0,1,2,3,4,5,6,7,8,9,10,11,12,13,14,15);
        }
#pragma unroll
        for (int mt = 0; mt < 2; ++mt)
#pragma unroll
            for (int nt = 0; nt < 4; ++nt)
                acc[mt][nt] = __builtin_amdgcn_wmma_f32_16x16x32_f16(
                    false, af[mt], false, bf[nt], (short)0, acc[mt][nt], false, false);
    }

    // epilogue: C/D layout M = v + 8*half, N = lane%16
#pragma unroll
    for (int mt = 0; mt < 2; ++mt) {
#pragma unroll
        for (int nt = 0; nt < 4; ++nt) {
            const int nIdx = nBlk + wn + nt*16 + l16;
            const float bv = bias[nIdx];
#pragma unroll
            for (int v = 0; v < 8; ++v) {
                const int mIdx = mBlk + wm + mt*16 + half*8 + v;
                float val = acc[mt][nt][v] + bv;
                if (act == 1) val = tanhf(val);
                C[(size_t)mIdx * N + nIdx] = val;
            }
        }
    }
}

// ---------------------------------------------------------------------------
// Embedding gather -> f16 activations
// ---------------------------------------------------------------------------
__global__ void gather_embed_f16(const int* __restrict__ ids,
                                 const float* __restrict__ emb,
                                 _Float16* __restrict__ x16)
{
    const int row = blockIdx.x;           // 0..B*T-1
    const int e   = threadIdx.x;          // 0..255
    x16[(size_t)row * E_ + e] = (_Float16)emb[(size_t)ids[row] * E_ + e];
}

__global__ void cvt_f32_to_f16(const float* __restrict__ in,
                               _Float16* __restrict__ out, int n)
{
    int i = blockIdx.x * blockDim.x + threadIdx.x;
    if (i < n) out[i] = (_Float16)in[i];
}

__global__ void bias_sum_k(const float* __restrict__ a,
                           const float* __restrict__ b,
                           float* __restrict__ o, int n)
{
    int i = blockIdx.x * blockDim.x + threadIdx.x;
    if (i < n) o[i] = a[i] + b[i];
}

// ---------------------------------------------------------------------------
// LSTM: one block per batch, 1024 threads (one per gate row).
// ---------------------------------------------------------------------------
__global__ __launch_bounds__(1024) void lstm_seq_kernel(
    const float*    __restrict__ xproj, // [B*T, 4H]
    const _Float16* __restrict__ Whh,   // [4H, H] f16
    float*          __restrict__ hout,  // [B*T, H] f32
    _Float16*       __restrict__ h16)   // [B*T, H] f16
{
    __shared__ _Float16 hbuf[H_];
    __shared__ float    gates[G_];

    const int b = blockIdx.x;
    const int g = threadIdx.x;          // 0..1023
    float c = 0.0f;
    if (g < H_) hbuf[g] = (_Float16)0.0f;
    __syncthreads();

    const _Float16* wrow = Whh + (size_t)g * H_;

    for (int t = 0; t < T_; ++t) {
        const size_t row = (size_t)b * T_ + t;
        float acc = xproj[row * G_ + g];
#pragma unroll 4
        for (int k = 0; k < H_; k += 8) {
            v8h w  = *(const v8h*)&wrow[k];
            v8h hv = *(const v8h*)&hbuf[k];
#pragma unroll
            for (int u = 0; u < 8; ++u) acc += (float)w[u] * (float)hv[u];
        }
        gates[g] = acc;
        __syncthreads();

        if (g < H_) {
            const float ig = 1.0f / (1.0f + __expf(-gates[g]));
            const float fg = 1.0f / (1.0f + __expf(-gates[H_ + g]));
            const float gg = tanhf(gates[2 * H_ + g]);
            const float og = 1.0f / (1.0f + __expf(-gates[3 * H_ + g]));
            c = fg * c + ig * gg;
            const float h = og * tanhf(c);
            hout[row * H_ + g] = h;
            h16 [row * H_ + g] = (_Float16)h;
            hbuf[g] = (_Float16)h;
        }
        __syncthreads();
    }
}

// ---------------------------------------------------------------------------
// Attention score: s[row] = dot(tanh_pre[row,:], a2_w) + a2_b. 1 wave / row.
// ---------------------------------------------------------------------------
__global__ __launch_bounds__(128) void attn_score_kernel(
    const float* __restrict__ t1,    // [B*T, H] (already tanh'd)
    const float* __restrict__ a2w,   // [H]
    const float* __restrict__ a2b,   // [1]
    float*       __restrict__ s)     // [B*T]
{
    const int lane = threadIdx.x & 31;
    const int wid  = threadIdx.x >> 5;
    const int row  = blockIdx.x * 4 + wid;
    const float* r = t1 + (size_t)row * H_;
    float acc = 0.0f;
#pragma unroll
    for (int k = lane; k < H_; k += 32) acc += r[k] * a2w[k];
#pragma unroll
    for (int off = 16; off > 0; off >>= 1) acc += __shfl_xor(acc, off, 32);
    if (lane == 0) s[row] = acc + a2b[0];
}

// ---------------------------------------------------------------------------
// Causal cumulative softmax pooling; emits combined=[ctx|h] in f16 for decoder.
// ---------------------------------------------------------------------------
__global__ __launch_bounds__(256) void cumctx_kernel(
    const float* __restrict__ s,     // [B*T]
    const float* __restrict__ hout,  // [B*T, H]
    _Float16*    __restrict__ comb)  // [B*T, 2H] f16
{
    __shared__ float red[256];
    const int b = blockIdx.x;
    const int j = threadIdx.x;
    const float* sb = s + (size_t)b * T_;

    float m = -1e30f;
    for (int t = j; t < T_; t += 256) m = fmaxf(m, sb[t]);
    red[j] = m;
    __syncthreads();
    for (int off = 128; off > 0; off >>= 1) {
        if (j < off) red[j] = fmaxf(red[j], red[j + off]);
        __syncthreads();
    }
    const float mx = red[0];

    float num = 0.0f, den = 0.0f;
    for (int t = 0; t < T_; ++t) {
        const float e = __expf(sb[t] - mx);
        den += e;
        const size_t row = (size_t)b * T_ + t;
        const float h = hout[row * H_ + j];
        num += e * h;
        const float ctx = num / den;
        comb[row * (size_t)K2_ + j]      = (_Float16)ctx;
        comb[row * (size_t)K2_ + H_ + j] = (_Float16)h;
    }
}

// ---------------------------------------------------------------------------
// Host launcher
// ---------------------------------------------------------------------------
extern "C" void kernel_launch(void* const* d_in, const int* in_sizes, int n_in,
                              void* d_out, int out_size, void* d_ws, size_t ws_size,
                              hipStream_t stream) {
    const int*   ids   = (const int*)  d_in[0];
    const float* emb   = (const float*)d_in[1];
    const float* W_ih  = (const float*)d_in[2];
    const float* W_hh  = (const float*)d_in[3];
    const float* b_ih  = (const float*)d_in[4];
    const float* b_hh  = (const float*)d_in[5];
    const float* a1_w  = (const float*)d_in[6];
    const float* a1_b  = (const float*)d_in[7];
    const float* a2_w  = (const float*)d_in[8];
    const float* a2_b  = (const float*)d_in[9];
    const float* dec_w = (const float*)d_in[10];
    const float* dec_b = (const float*)d_in[11];
    float* out = (float*)d_out;

    char* ws = (char*)d_ws;
    size_t off = 0;
    auto alloc = [&](size_t bytes) { char* p = ws + off; off = (off + bytes + 255) & ~(size_t)255; return p; };

    _Float16* decw16 = (_Float16*)alloc((size_t)V_ * K2_ * 2);      // 32.75 MB
    float*    xproj  = (float*)   alloc((size_t)B_ * T_ * G_ * 4);  // 16 MB (reused as t1)
    float*    houtf  = (float*)   alloc((size_t)B_ * T_ * H_ * 4);
    _Float16* x16    = (_Float16*)alloc((size_t)B_ * T_ * E_ * 2);
    _Float16* h16    = (_Float16*)alloc((size_t)B_ * T_ * H_ * 2);
    _Float16* comb16 = (_Float16*)alloc((size_t)B_ * T_ * K2_ * 2);
    _Float16* wih16  = (_Float16*)alloc((size_t)G_ * E_ * 2);
    _Float16* whh16  = (_Float16*)alloc((size_t)G_ * H_ * 2);
    _Float16* a1w16  = (_Float16*)alloc((size_t)H_ * H_ * 2);
    float*    bsum   = (float*)   alloc((size_t)G_ * 4);
    float*    sbuf   = (float*)   alloc((size_t)B_ * T_ * 4);
    float*    t1     = xproj;   // alias: xproj dead after LSTM
    (void)in_sizes; (void)n_in; (void)out_size; (void)ws_size;

    bias_sum_k<<<dim3(4), dim3(256), 0, stream>>>(b_ih, b_hh, bsum, G_);
    cvt_f32_to_f16<<<dim3((G_ * E_ + 255) / 256), dim3(256), 0, stream>>>(W_ih, wih16, G_ * E_);
    cvt_f32_to_f16<<<dim3((G_ * H_ + 255) / 256), dim3(256), 0, stream>>>(W_hh, whh16, G_ * H_);
    cvt_f32_to_f16<<<dim3((H_ * H_ + 255) / 256), dim3(256), 0, stream>>>(a1_w, a1w16, H_ * H_);
    cvt_f32_to_f16<<<dim3((V_ * K2_ + 255) / 256), dim3(256), 0, stream>>>(dec_w, decw16, V_ * K2_);

    gather_embed_f16<<<dim3(B_ * T_), dim3(E_), 0, stream>>>(ids, emb, x16);

    wmma_gemm_f16<<<dim3(G_ / 128, (B_ * T_) / 128), dim3(256), 0, stream>>>(
        x16, wih16, bsum, xproj, B_ * T_, G_, E_, 0);

    lstm_seq_kernel<<<dim3(B_), dim3(G_), 0, stream>>>(xproj, whh16, houtf, h16);

    wmma_gemm_f16<<<dim3(H_ / 128, (B_ * T_) / 128), dim3(256), 0, stream>>>(
        h16, a1w16, a1_b, t1, B_ * T_, H_, H_, 1);

    attn_score_kernel<<<dim3((B_ * T_) / 4), dim3(128), 0, stream>>>(t1, a2_w, a2_b, sbuf);

    cumctx_kernel<<<dim3(B_), dim3(256), 0, stream>>>(sbuf, houtf, comb16);

    wmma_gemm_f16<<<dim3(V_ / 128, (B_ * T_) / 128), dim3(256), 0, stream>>>(
        comb16, decw16, dec_b, out, B_ * T_, V_, K2_, 0);
}
